// Attention_2525440770966
// MI455X (gfx1250) — matrix operand order repaired and checked by reading, not compiled, not run on hardware
//
#include <hip/hip_runtime.h>
#include <stdint.h>

typedef __attribute__((ext_vector_type(16))) _Float16 v16h;
typedef __attribute__((ext_vector_type(8)))  float    v8f;

union Frag16 {
  v16h     v;
  uint4    q[2];
  _Float16 h[16];
  uint32_t u[8];
};

// max-reduce across 16-lane half-wave groups using DPP (pure VALU, no LDS)
__device__ __forceinline__ float rowmax16(float x) {
  // xor1: quad_perm(1,0,3,2)
  x = fmaxf(x, __int_as_float(__builtin_amdgcn_mov_dpp(__float_as_int(x), 0xB1, 0xF, 0xF, true)));
  // xor2: quad_perm(2,3,0,1)
  x = fmaxf(x, __int_as_float(__builtin_amdgcn_mov_dpp(__float_as_int(x), 0x4E, 0xF, 0xF, true)));
  // quads uniform -> row_half_mirror acts as xor4
  x = fmaxf(x, __int_as_float(__builtin_amdgcn_mov_dpp(__float_as_int(x), 0x141, 0xF, 0xF, true)));
  // 8-groups uniform -> row_mirror acts as xor8 (stays within 16-lane row)
  x = fmaxf(x, __int_as_float(__builtin_amdgcn_mov_dpp(__float_as_int(x), 0x140, 0xF, 0xF, true)));
  return x;
}

__device__ __forceinline__ float fexp2(float x) { return __builtin_amdgcn_exp2f(x); }

#define BQ 4
#define NN 2048
#define CC 1024
#define HH 16
#define DD 64
#define BM 128   // query rows per workgroup (8 waves x 16)
#define BN 64    // keys per inner step (two WMMA K-chunks for PV)

__global__ __launch_bounds__(256)
void fa_fwd_f16_wmma(const float* __restrict__ Qp,
                     const float* __restrict__ Kp,
                     const float* __restrict__ Vp,
                     float* __restrict__ Op) {
  // Double-buffered K tile: 64 keys x 64 dims (padded row = 72 f16 = 144B)
  __shared__ __attribute__((aligned(16))) _Float16 sK[2][BN][72];
  // Double-buffered V tile TRANSPOSED: 64 dims x 64 keys (row = 144B)
  __shared__ __attribute__((aligned(16))) _Float16 sV[2][DD][72];
  // per-wave P staging for C-layout -> A-layout transpose (row = 128B)
  __shared__ __attribute__((aligned(16))) _Float16 sP[8][16][64];

  const int tid  = threadIdx.x;
  const int w    = tid >> 5;          // wave id 0..7
  const int lane = tid & 31;
  const int bh   = blockIdx.y;
  const int b    = bh >> 4;           // / HH
  const int h    = bh & 15;
  const int m0   = blockIdx.x * BM + w * 16;

  const int nlo = lane & 15;
  const int hi  = (lane & 16) ? 1 : 0;
  const int dcol = tid & 63;
  const int grp  = tid >> 6;

  // ---- Q A-fragments (two K=32 chunks), pre-scaled by log2(e)/sqrt(d) ----
  // exp2(s*log2e - m2) == exp(s - m): softmax unchanged, saves a mul per exp.
  Frag16 aQ[2];
  {
    const int    m    = m0 + nlo;                       // A-layout: lane holds row M = lane&15
    const float* qrow = Qp + ((size_t)(b * NN + m)) * CC + h * DD;
    const float  sc   = 0.125f * 1.44269504088896f;     // log2(e)/sqrt(64)
#pragma unroll
    for (int c = 0; c < 2; ++c) {
      const int k0 = c * 32 + (hi ? 8 : 0);             // halves 0..7 -> k0.., 8..15 -> k0+16..
      float4 f0 = *(const float4*)(qrow + k0);
      float4 f1 = *(const float4*)(qrow + k0 + 4);
      float4 f2 = *(const float4*)(qrow + k0 + 16);
      float4 f3 = *(const float4*)(qrow + k0 + 20);
      const float* p0 = &f0.x; const float* p1 = &f1.x;
      const float* p2 = &f2.x; const float* p3 = &f3.x;
#pragma unroll
      for (int i = 0; i < 4; ++i) {
        aQ[c].h[i]      = (_Float16)(p0[i] * sc);
        aQ[c].h[4 + i]  = (_Float16)(p1[i] * sc);
        aQ[c].h[8 + i]  = (_Float16)(p2[i] * sc);
        aQ[c].h[12 + i] = (_Float16)(p3[i] * sc);
      }
    }
  }

  // B-fragment of all-ones (f16 1.0 = 0x3C00) for the row-sum matmul trick
  Frag16 ones;
#pragma unroll
  for (int i = 0; i < 8; ++i) ones.u[i] = 0x3C003C00u;

  v8f acc[4];
  v8f lacc;   // running softmax denominator per row, via lacc = wmma(P, ones, lacc)
  {
    v8f z = {0.f,0.f,0.f,0.f,0.f,0.f,0.f,0.f};
    acc[0] = z; acc[1] = z; acc[2] = z; acc[3] = z; lacc = z;
  }
  float mrow[8];
#pragma unroll
  for (int j = 0; j < 8; ++j) mrow[j] = -INFINITY;

  // ---- prologue: stage tile 0 into buffer 0 ----
  {
    const float* kb = Kp + ((size_t)(b * NN)) * CC + h * DD;
    const float* vb = Vp + ((size_t)(b * NN)) * CC + h * DD;
#pragma unroll
    for (int e0 = 0; e0 < 4; ++e0) {
      const int e   = tid + e0 * 256;
      const int row = e >> 4;
      const int c4  = (e & 15) * 4;
      float4 f = *(const float4*)(kb + (size_t)row * CC + c4);
      const float* fp = &f.x;
#pragma unroll
      for (int i = 0; i < 4; ++i) sK[0][row][c4 + i] = (_Float16)fp[i];
    }
#pragma unroll
    for (int r = 0; r < 16; ++r)
      sV[0][dcol][grp * 16 + r] = (_Float16)vb[(size_t)(grp * 16 + r) * CC + dcol];
  }
  __syncthreads();

  float4 kreg[4];
  float  vreg[16];

  for (int j0 = 0, p = 0; ; j0 += BN, p ^= 1) {
    const bool more = (j0 + BN) < NN;

    // ---- issue next tile's global loads now; latency overlaps the WMMA body ----
    if (more) {
      const float* kb = Kp + ((size_t)(b * NN + j0 + BN)) * CC + h * DD;
      const float* vb = Vp + ((size_t)(b * NN + j0 + BN)) * CC + h * DD;
#pragma unroll
      for (int e0 = 0; e0 < 4; ++e0) {
        const int e   = tid + e0 * 256;
        const int row = e >> 4;
        const int c4  = (e & 15) * 4;
        kreg[e0] = *(const float4*)(kb + (size_t)row * CC + c4);
      }
#pragma unroll
      for (int r = 0; r < 16; ++r)
        vreg[r] = vb[(size_t)(grp * 16 + r) * CC + dcol];
      if (j0 + 2 * BN < NN) {   // pull tile j+2 toward near caches
        __builtin_prefetch(kb + (size_t)BN * CC + dcol, 0, 3);
        __builtin_prefetch(vb + (size_t)BN * CC + dcol, 0, 3);
      }
    }

    // ---- S = (Q*scale) * K^T : four 16x16 tiles, K-dim 64 chained as 2x32 ----
    v8f st[4];
#pragma unroll
    for (int t = 0; t < 4; ++t) {
      v8f s = {0.f,0.f,0.f,0.f,0.f,0.f,0.f,0.f};
#pragma unroll
      for (int c = 0; c < 2; ++c) {
        Frag16 bK;   // B-layout: lane = column n (key), halves 0..7 -> k 8h.., 8..15 -> k 16+8h..
        const _Float16* kp = &sK[p][t * 16 + nlo][c * 32 + (hi ? 8 : 0)];
        bK.q[0] = *(const uint4*)kp;
        bK.q[1] = *(const uint4*)(kp + 16);
        s = __builtin_amdgcn_wmma_f32_16x16x32_f16(false, aQ[c].v, false, bK.v,
                                                   (short)0, s, false, false);
      }
      st[t] = s;
    }

    // ---- online softmax (exp2 domain): row max via DPP, rescale, P -> LDS ----
#pragma unroll
    for (int j = 0; j < 8; ++j) {
      const float t01  = fmaxf(st[0][j], st[1][j]);
      const float t23  = fmaxf(st[2][j], st[3][j]);
      const float tmax = rowmax16(fmaxf(t01, t23));
      const float mnew = fmaxf(mrow[j], tmax);
      const float corr = fexp2(mrow[j] - mnew);
      mrow[j] = mnew;
      lacc[j] *= corr;
#pragma unroll
      for (int a = 0; a < 4; ++a) acc[a][j] *= corr;
      const int mloc = j + (hi ? 8 : 0);
#pragma unroll
      for (int t = 0; t < 4; ++t)
        sP[w][mloc][t * 16 + nlo] = (_Float16)fexp2(st[t][j] - mnew);
    }
    asm volatile("s_wait_dscnt 0x0" ::: "memory");   // same-wave LDS is in-order

    // ---- O += P * V (4 d-tiles x 2 K-chunks) and l += P * ones ----
    Frag16 aP[2];
#pragma unroll
    for (int c = 0; c < 2; ++c) {
      const _Float16* pp = &sP[w][nlo][c * 32 + (hi ? 8 : 0)];
      aP[c].q[0] = *(const uint4*)pp;
      aP[c].q[1] = *(const uint4*)(pp + 16);
    }
#pragma unroll
    for (int a = 0; a < 4; ++a) {
#pragma unroll
      for (int c = 0; c < 2; ++c) {
        Frag16 bV;   // lane = column d, halves 0..7 -> keys 8h.., 8..15 -> keys 16+8h.. (V transposed in LDS)
        const _Float16* vp = &sV[p][a * 16 + nlo][c * 32 + (hi ? 8 : 0)];
        bV.q[0] = *(const uint4*)vp;
        bV.q[1] = *(const uint4*)(vp + 16);
        acc[a] = __builtin_amdgcn_wmma_f32_16x16x32_f16(false, aP[c].v, false, bV.v,
                                                        (short)0, acc[a], false, false);
      }
    }
    lacc = __builtin_amdgcn_wmma_f32_16x16x32_f16(false, aP[0].v, false, ones.v,
                                                  (short)0, lacc, false, false);
    lacc = __builtin_amdgcn_wmma_f32_16x16x32_f16(false, aP[1].v, false, ones.v,
                                                  (short)0, lacc, false, false);

    if (!more) break;

    // ---- drain next tile from registers into the alternate buffer ----
    __syncthreads();   // all reads of sK/sV[p^1] (from iter j-1) are done
#pragma unroll
    for (int e0 = 0; e0 < 4; ++e0) {
      const int e   = tid + e0 * 256;
      const int row = e >> 4;
      const int c4  = (e & 15) * 4;
      const float* fp = &kreg[e0].x;
#pragma unroll
      for (int i = 0; i < 4; ++i) sK[p ^ 1][row][c4 + i] = (_Float16)fp[i];
    }
#pragma unroll
    for (int r = 0; r < 16; ++r)
      sV[p ^ 1][dcol][grp * 16 + r] = (_Float16)vreg[r];
    __syncthreads();   // next iteration may read buffer p^1
  }

  // ---- epilogue: O / l ----
#pragma unroll
  for (int j = 0; j < 8; ++j) {
    const float inv  = __builtin_amdgcn_rcpf(lacc[j]);
    const int   mloc = j + (hi ? 8 : 0);
    float* orow = Op + ((size_t)(b * NN + m0 + mloc)) * CC + h * DD;
#pragma unroll
    for (int a = 0; a < 4; ++a)
      orow[a * 16 + nlo] = acc[a][j] * inv;
  }
}

extern "C" void kernel_launch(void* const* d_in, const int* in_sizes, int n_in,
                              void* d_out, int out_size, void* d_ws, size_t ws_size,
                              hipStream_t stream) {
  (void)in_sizes; (void)n_in; (void)out_size; (void)d_ws; (void)ws_size;
  const float* q = (const float*)d_in[0];
  const float* k = (const float*)d_in[1];
  const float* v = (const float*)d_in[2];
  float* out = (float*)d_out;
  dim3 grid(NN / BM, BQ * HH);   // 16 x 64
  dim3 block(256);               // 8 waves
  fa_fwd_f16_wmma<<<grid, block, 0, stream>>>(q, k, v, out);
}
